// RecurrentSpikingNetwork_56538949485334
// MI455X (gfx1250) — compile-verified, matrix operand order checked by reading, not run on hardware
//
#include <hip/hip_runtime.h>
#include <hip/hip_bf16.h>

// ---------------------------------------------------------------------------
// Recurrent adaptive-LIF SNN forward for MI455X (gfx1250), persistent-kernel
// design:
//  * single kernel runs all 200 recurrent steps; 16 blocks sync via a
//    device-scope sense-reversing grid barrier (2 barriers/step) instead of
//    600 kernel launches.
//  * each block owns a fixed 64-column N-strip; its weight slices
//    (Wd/Wr: 2x130KB padded + W1: 17KB) live in the 320KB WGP LDS for the
//    whole run -> weights cross HBM/L2 exactly once.
//  * weight staging uses the Tensor Data Mover (tensor_load_to_lds, D#
//    descriptors with LDS padding for bank spread) when the builtin exists,
//    else a manual copy producing the identical layout.
//  * GEMMs use v_wmma_f32_16x16x32_f16: spike activations are exact in f16,
//    weights f16, accumulation f32.
// ---------------------------------------------------------------------------

typedef __attribute__((ext_vector_type(16))) _Float16 v16h;
typedef __attribute__((ext_vector_type(8)))  _Float16 v8h;
typedef __attribute__((ext_vector_type(8)))  float    v8f;
typedef unsigned int u32x4 __attribute__((ext_vector_type(4)));
typedef int          i32x8 __attribute__((ext_vector_type(8)));
typedef int          i32x4 __attribute__((ext_vector_type(4)));

union F16x16 { v16h v; v8h p[2]; };

#define BATCH 128
#define TSEQ  200
#define DIN   120
#define DPAD  128
#define HID   1024
#define NOUT  12
#define B_J0  0.01f
#define BETA  1.8f
#define NBLK  16          // persistent blocks; one 64-col N-strip each

// LDS layout (halfs). Wd/Wr rows: 1024 halfs + 16B pad per 256 dwords
// (TDM pad_interval=7, pad_amount=4 dwords) -> stride 1040 halfs, second
// half-row (k>=512) shifted by +8 halfs; 16B alignment kept everywhere.
#define WSTRIDE 1040
#define W1STRIDE 136      // 128 halfs + 16B pad (interval=5, amount=4 dwords)
#define LDS_WD_OFF 0
#define LDS_WR_OFF (64 * WSTRIDE)
#define LDS_W1_OFF (128 * WSTRIDE)
#define LDS_HALFS  (128 * WSTRIDE + 64 * W1STRIDE)
#define LDS_BYTES  (LDS_HALFS * 2)

#if defined(__gfx1250__) && __has_builtin(__builtin_amdgcn_tensor_load_to_lds)
#define SNN_TDM 1
#endif

// Load a 16-half WMMA operand fragment (A or B) for this lane.
// lane<16 -> K offsets {0..7,16..23}; lane>=16 -> {8..15,24..31}.
__device__ __forceinline__ v16h load_frag(const _Float16* rowbase, int half) {
    F16x16 f;
    f.p[0] = *(const v8h*)(rowbase + 8 * half);
    f.p[1] = *(const v8h*)(rowbase + 16 + 8 * half);
    return f.v;
}

// ------------------------- one-time prep kernels ---------------------------

__global__ void cvt_w1_k(const float* __restrict__ W1, _Float16* __restrict__ W1h) {
    int idx = blockIdx.x * blockDim.x + threadIdx.x;     // 1024*128
    int j = idx >> 7, k = idx & 127;
    W1h[idx] = (_Float16)(k < DIN ? W1[j * DIN + k] : 0.0f);
}

__global__ void cvt_w_k(const float* __restrict__ W, _Float16* __restrict__ Wh) {
    int idx = blockIdx.x * blockDim.x + threadIdx.x;     // 1024*1024
    Wh[idx] = (_Float16)W[idx];
}

__global__ void decays_k(const float* tm1, const float* ta1,
                         const float* tmr, const float* tar, const float* tm2,
                         float* a1, float* r1, float* ar, float* rr, float* a2) {
    int j = blockIdx.x * blockDim.x + threadIdx.x;
    if (j < HID) {
        a1[j] = expf(-1.0f / tm1[j]);
        r1[j] = expf(-1.0f / ta1[j]);
        ar[j] = expf(-1.0f / tmr[j]);
        rr[j] = expf(-1.0f / tar[j]);
    }
    if (j < NOUT) a2[j] = expf(-1.0f / tm2[j]);
}

// ternary input spikes xs[t][b][k] in {-1,0,1}, K padded 120->128
__global__ void spikes_k(const float* __restrict__ inputs, const float* __restrict__ thr_p,
                         _Float16* __restrict__ xs) {
    int idx = blockIdx.x * blockDim.x + threadIdx.x;     // T*B*128
    int k = idx & 127;
    int b = (idx >> 7) & 127;
    int t = idx >> 14;
    float v = 0.0f;
    if (k < DIN) {
        int c = k / 40, f = k % 40;
        float x   = inputs[((b * 3 + c) * TSEQ + t) * 40 + f];
        float thr = *thr_p;
        v = (x > thr ? 1.0f : 0.0f) - (x < -thr ? 1.0f : 0.0f);
    }
    xs[idx] = (_Float16)v;
}

__global__ void init_state_k(float* mem1, float* b1s, _Float16* s1,
                             float* memr, float* bbr, _Float16* sr0, _Float16* sr1,
                             float* mem2, float* acc, unsigned* bar) {
    int i = blockIdx.x * blockDim.x + threadIdx.x;       // 128*1024
    mem1[i] = 0.0f;  b1s[i] = B_J0;  s1[i]  = (_Float16)0.0f;
    memr[i] = 0.0f;  bbr[i] = B_J0;  sr0[i] = (_Float16)0.0f;  sr1[i] = (_Float16)0.0f;
    if (i < BATCH * NOUT) { mem2[i] = 0.0f; acc[i] = 0.0f; }
    if (i < 2) bar[i] = 0u;
}

// --------------------------- persistent kernel -----------------------------

extern __shared__ _Float16 g_lds[];

__device__ __forceinline__ const _Float16* wrow(const _Float16* base, int r, int kc) {
    return base + r * WSTRIDE + kc + (kc >= 512 ? 8 : 0);
}

__device__ inline void grid_sync(unsigned* cnt, unsigned* gen) {
    __threadfence();
    __syncthreads();
    if (threadIdx.x == 0) {
        unsigned g = __hip_atomic_load(gen, __ATOMIC_RELAXED, __HIP_MEMORY_SCOPE_AGENT);
        if (__hip_atomic_fetch_add(cnt, 1u, __ATOMIC_ACQ_REL, __HIP_MEMORY_SCOPE_AGENT)
            == NBLK - 1) {
            __hip_atomic_store(cnt, 0u, __ATOMIC_RELAXED, __HIP_MEMORY_SCOPE_AGENT);
            __hip_atomic_store(gen, g + 1u, __ATOMIC_RELEASE, __HIP_MEMORY_SCOPE_AGENT);
        } else {
            while (__hip_atomic_load(gen, __ATOMIC_ACQUIRE, __HIP_MEMORY_SCOPE_AGENT) == g) {
                __builtin_amdgcn_s_sleep(2);
            }
        }
    }
    __syncthreads();
    __threadfence();
}

#if defined(SNN_TDM)
// Issue one TDM 2D tile load: rows x tile_k f16 elements, row stride
// stride_elems, into LDS at lds_off with padding config padcfg (group1 dword0
// bits 20..31). D# bit layout per CDNA5 ISA ch.8 (CDNA5 ISA).
__device__ __forceinline__ void tdm_load_2d(const _Float16* gsrc, unsigned lds_off,
                                            unsigned tile_k, unsigned rows,
                                            unsigned stride_elems, unsigned padcfg) {
    unsigned long long ga = (unsigned long long)(uintptr_t)gsrc;
    u32x4 g0;
    g0.x = 1u;                                        // count=1 (valid user D#)
    g0.y = lds_off;                                   // lds_addr (bytes)
    g0.z = (unsigned)(ga & 0xffffffffu);              // global_addr[31:0]
    g0.w = (unsigned)((ga >> 32) & 0x01ffffffu) | (2u << 30);  // addr[56:32] | type=2
    i32x8 g1;
    g1[0] = (int)((1u << 16) | padcfg);               // data_size=2B | pad cfg
    g1[1] = (int)((tile_k & 0xffffu) << 16);          // tensor_dim0[15:0]
    g1[2] = (int)((tile_k >> 16) | ((rows & 0xffffu) << 16));   // td0 hi | td1 lo
    g1[3] = (int)((rows >> 16) | ((tile_k & 0xffffu) << 16));   // td1 hi | tile_dim0
    g1[4] = (int)(rows & 0xffffu);                    // tile_dim1 (tile_dim2=0)
    g1[5] = (int)stride_elems;                        // tensor_dim0_stride lo32
    g1[6] = 0; g1[7] = 0;
    i32x4 gz = {0, 0, 0, 0};
#if __clang_major__ >= 23
    i32x8 gz8 = {0, 0, 0, 0, 0, 0, 0, 0};
    __builtin_amdgcn_tensor_load_to_lds(g0, g1, gz, gz, gz8, 0);
#else
    __builtin_amdgcn_tensor_load_to_lds(g0, g1, gz, gz, 0);
#endif
}
#endif

// manual staging fallback producing the identical padded LDS layout
__device__ void stage_rows(const _Float16* __restrict__ gsrc, _Float16* dst,
                           int cols, int stride_lds) {
    const int vecs = cols >> 3;                       // v8h per row
    for (int i = threadIdx.x; i < 64 * vecs; i += 256) {
        int r = i / vecs, k = (i % vecs) << 3;
        v8h v = *(const v8h*)(gsrc + (size_t)r * cols + k);
        *(v8h*)(dst + r * stride_lds + k + ((cols == HID && k >= 512) ? 8 : 0)) = v;
    }
}

__global__ void __launch_bounds__(256, 1)
snn_persist_k(const _Float16* __restrict__ xs,       // [T,128,128]
              const _Float16* __restrict__ W1h,      // [1024,128]
              const _Float16* __restrict__ Wdh,      // [1024,1024]
              const _Float16* __restrict__ Wrh,      // [1024,1024]
              const float* __restrict__ b1,  const float* __restrict__ bd,
              const float* __restrict__ br,  const float* __restrict__ W2,
              const float* __restrict__ b2,
              const float* __restrict__ alpha1, const float* __restrict__ ro1,
              const float* __restrict__ alphar, const float* __restrict__ ror,
              const float* __restrict__ alpha2,
              float* __restrict__ mem1, float* __restrict__ b1s,
              _Float16* __restrict__ s1,
              float* __restrict__ memr, float* __restrict__ bbr,
              _Float16* __restrict__ srA, _Float16* __restrict__ srB,
              float* __restrict__ mem2, float* __restrict__ accum,
              float* __restrict__ out, unsigned* __restrict__ bar)
{
    _Float16* sWd = g_lds + LDS_WD_OFF;
    _Float16* sWr = g_lds + LDS_WR_OFF;
    _Float16* sW1 = g_lds + LDS_W1_OFF;
    const int n0 = blockIdx.x * 64;

    // ---- stage this block's weight slices into LDS (resident for all steps)
#if defined(SNN_TDM)
    if (threadIdx.x < 32) {
        const unsigned ldsbase = (unsigned)(uintptr_t)g_lds;
        const unsigned padW  = (1u << 20) | (7u << 22) | (3u << 25); // 16B per 256dw
        const unsigned padW1 = (1u << 20) | (5u << 22) | (3u << 25); // 16B per 64dw
        tdm_load_2d(Wdh + (size_t)n0 * HID, ldsbase + LDS_WD_OFF * 2, HID, 64, HID, padW);
        tdm_load_2d(Wrh + (size_t)n0 * HID, ldsbase + LDS_WR_OFF * 2, HID, 64, HID, padW);
        tdm_load_2d(W1h + (size_t)n0 * DPAD, ldsbase + LDS_W1_OFF * 2, DPAD, 64, DPAD, padW1);
        __builtin_amdgcn_s_wait_tensorcnt(0);
    }
#else
    stage_rows(Wdh + (size_t)n0 * HID, sWd, HID, WSTRIDE);
    stage_rows(Wrh + (size_t)n0 * HID, sWr, HID, WSTRIDE);
    stage_rows(W1h + (size_t)n0 * DPAD, sW1, DPAD, W1STRIDE);
#endif
    __syncthreads();

    const int wave = threadIdx.x >> 5;
    const int lane = threadIdx.x & 31;
    const int half = lane >> 4, lid = lane & 15;
    const int m0 = wave << 4;                 // wave owns one 16-row M strip

    unsigned* bcnt = bar;
    unsigned* bgen = bar + 1;

    for (int t = 0; t < TSEQ; ++t) {
        const _Float16* xs_t = xs + (size_t)t * BATCH * DPAD;
        _Float16* sr_rd = (t & 1) ? srB : srA;
        _Float16* sr_wr = (t & 1) ? srA : srB;

        // ---------------- phase 1: layer 1 (K=128) + LIF --------------------
        {
            v8f acc[4] = {{}, {}, {}, {}};
            for (int kc = 0; kc < DPAD; kc += 32) {
                v16h a = load_frag(xs_t + (m0 + lid) * DPAD + kc, half);
#pragma unroll
                for (int tn = 0; tn < 4; ++tn) {
                    v16h bf = load_frag(sW1 + (tn * 16 + lid) * W1STRIDE + kc, half);
                    acc[tn] = __builtin_amdgcn_wmma_f32_16x16x32_f16(
                        false, a, false, bf, (short)0, acc[tn], false, false);
                }
            }
#pragma unroll
            for (int tn = 0; tn < 4; ++tn) {
                const int j = n0 + tn * 16 + lid;
                const float al = alpha1[j], ro = ro1[j], bias = b1[j];
#pragma unroll
                for (int v = 0; v < 8; ++v) {
                    const int idx = (m0 + v + 8 * half) * HID + j;
                    float x   = acc[tn][v] + bias;
                    float spk = (float)s1[idx];
                    float bb  = ro * b1s[idx] + (1.0f - ro) * spk;
                    float Bth = B_J0 + BETA * bb;
                    float mm  = mem1[idx] * al + (1.0f - al) * x - Bth * spk;
                    mem1[idx] = mm;
                    b1s[idx]  = bb;
                    s1[idx]   = (_Float16)((mm - Bth) > 0.0f ? 1.0f : 0.0f);
                }
            }
        }
        grid_sync(bcnt, bgen);          // all s1 written before K=1024 reads

        // ------- phase 2: recurrent layer, s1@Wd^T + sr@Wr^T (K=1024) -------
        {
            v8f acc[4] = {{}, {}, {}, {}};
            for (int kc = 0; kc < HID; kc += 32) {
                v16h a = load_frag(s1 + (m0 + lid) * HID + kc, half);
#pragma unroll
                for (int tn = 0; tn < 4; ++tn) {
                    v16h bf = load_frag(wrow(sWd, tn * 16 + lid, kc), half);
                    acc[tn] = __builtin_amdgcn_wmma_f32_16x16x32_f16(
                        false, a, false, bf, (short)0, acc[tn], false, false);
                }
            }
            for (int kc = 0; kc < HID; kc += 32) {
                v16h a = load_frag(sr_rd + (m0 + lid) * HID + kc, half);
#pragma unroll
                for (int tn = 0; tn < 4; ++tn) {
                    v16h bf = load_frag(wrow(sWr, tn * 16 + lid, kc), half);
                    acc[tn] = __builtin_amdgcn_wmma_f32_16x16x32_f16(
                        false, a, false, bf, (short)0, acc[tn], false, false);
                }
            }
#pragma unroll
            for (int tn = 0; tn < 4; ++tn) {
                const int j = n0 + tn * 16 + lid;
                const float al = alphar[j], ro = ror[j], bias = bd[j] + br[j];
#pragma unroll
                for (int v = 0; v < 8; ++v) {
                    const int idx = (m0 + v + 8 * half) * HID + j;
                    float x   = acc[tn][v] + bias;
                    float spk = (float)sr_rd[idx];
                    float bb  = ro * bbr[idx] + (1.0f - ro) * spk;
                    float Bth = B_J0 + BETA * bb;
                    float mm  = memr[idx] * al + (1.0f - al) * x - Bth * spk;
                    memr[idx] = mm;
                    bbr[idx]  = bb;
                    sr_wr[idx] = (_Float16)((mm - Bth) > 0.0f ? 1.0f : 0.0f);
                }
            }
        }
        grid_sync(bcnt, bgen);          // all sr_wr written before readout

        // ---------------- phase 3: readout (one wave per batch row) ---------
        {
            const int b = blockIdx.x * 8 + wave;
            const _Float16* srow = sr_wr + (size_t)b * HID;
            float sums[NOUT];
#pragma unroll
            for (int o = 0; o < NOUT; ++o) sums[o] = 0.0f;
            for (int k = lane; k < HID; k += 32) {
                float s = (float)srow[k];
#pragma unroll
                for (int o = 0; o < NOUT; ++o) sums[o] += s * W2[o * HID + k];
            }
#pragma unroll
            for (int o = 0; o < NOUT; ++o) {
                float v = sums[o];
#pragma unroll
                for (int off = 16; off > 0; off >>= 1) v += __shfl_xor(v, off, 32);
                if (lane == 0) {
                    const float al = alpha2[o];
                    float m = mem2[b * NOUT + o] * al + (1.0f - al) * (v + b2[o]);
                    mem2[b * NOUT + o]  = m;
                    accum[b * NOUT + o] += m;
                }
            }
        }
        // no barrier needed: phase 1(t+1) touches disjoint state, and the
        // phase-1 barrier orders phase 3 reads before phase 2(t+1) writes.
    }

    grid_sync(bcnt, bgen);
    if (blockIdx.x == 0 && threadIdx.x < BATCH) {
        const int b = threadIdx.x;
        float a[NOUT], mx = -1e30f;
#pragma unroll
        for (int o = 0; o < NOUT; ++o) {
            a[o] = accum[b * NOUT + o] * (1.0f / (float)TSEQ);
            mx = fmaxf(mx, a[o]);
        }
        float s = 0.0f;
#pragma unroll
        for (int o = 0; o < NOUT; ++o) s += expf(a[o] - mx);
        float l = logf(s);
#pragma unroll
        for (int o = 0; o < NOUT; ++o) out[b * NOUT + o] = a[o] - mx - l;
    }
}

// ------------------------------- host side ---------------------------------

extern "C" void kernel_launch(void* const* d_in, const int* in_sizes, int n_in,
                              void* d_out, int out_size, void* d_ws, size_t ws_size,
                              hipStream_t stream) {
    (void)in_sizes; (void)n_in; (void)out_size; (void)ws_size;
    const float* inputs   = (const float*)d_in[0];
    const float* thr      = (const float*)d_in[1];
    const float* W1       = (const float*)d_in[2];
    const float* b1       = (const float*)d_in[3];
    const float* Wd       = (const float*)d_in[4];
    const float* bd       = (const float*)d_in[5];
    const float* Wr       = (const float*)d_in[6];
    const float* br       = (const float*)d_in[7];
    const float* W2       = (const float*)d_in[8];
    const float* b2       = (const float*)d_in[9];
    const float* tau_m1   = (const float*)d_in[10];
    const float* tau_adp1 = (const float*)d_in[11];
    const float* tau_m_r  = (const float*)d_in[12];
    const float* tau_adp_r= (const float*)d_in[13];
    const float* tau_m2   = (const float*)d_in[14];
    float* out = (float*)d_out;

    char* ws = (char*)d_ws;
    size_t off = 0;
    auto alloc = [&](size_t bytes) -> char* {
        char* p = ws + off;
        off = (off + bytes + 255) & ~((size_t)255);
        return p;
    };
    _Float16* W1h   = (_Float16*)alloc((size_t)HID * DPAD * 2);
    _Float16* Wdh   = (_Float16*)alloc((size_t)HID * HID * 2);
    _Float16* Wrh   = (_Float16*)alloc((size_t)HID * HID * 2);
    _Float16* xs    = (_Float16*)alloc((size_t)TSEQ * BATCH * DPAD * 2);
    _Float16* s1    = (_Float16*)alloc((size_t)BATCH * HID * 2);
    _Float16* srA   = (_Float16*)alloc((size_t)BATCH * HID * 2);
    _Float16* srB   = (_Float16*)alloc((size_t)BATCH * HID * 2);
    float* mem1     = (float*)alloc((size_t)BATCH * HID * 4);
    float* b1s      = (float*)alloc((size_t)BATCH * HID * 4);
    float* memr     = (float*)alloc((size_t)BATCH * HID * 4);
    float* bbr      = (float*)alloc((size_t)BATCH * HID * 4);
    float* mem2     = (float*)alloc((size_t)BATCH * NOUT * 4);
    float* accum    = (float*)alloc((size_t)BATCH * NOUT * 4);
    float* alpha1   = (float*)alloc(HID * 4);
    float* ro1      = (float*)alloc(HID * 4);
    float* alphar   = (float*)alloc(HID * 4);
    float* ror      = (float*)alloc(HID * 4);
    float* alpha2   = (float*)alloc(NOUT * 4);
    unsigned* bar   = (unsigned*)alloc(2 * sizeof(unsigned));

    cvt_w1_k<<<(HID * DPAD) / 256, 256, 0, stream>>>(W1, W1h);
    cvt_w_k <<<(HID * HID) / 256, 256, 0, stream>>>(Wd, Wdh);
    cvt_w_k <<<(HID * HID) / 256, 256, 0, stream>>>(Wr, Wrh);
    decays_k<<<HID / 256, 256, 0, stream>>>(tau_m1, tau_adp1, tau_m_r, tau_adp_r, tau_m2,
                                            alpha1, ro1, alphar, ror, alpha2);
    spikes_k<<<(TSEQ * BATCH * DPAD) / 256, 256, 0, stream>>>(inputs, thr, xs);
    init_state_k<<<(BATCH * HID) / 256, 256, 0, stream>>>(mem1, b1s, s1, memr, bbr,
                                                          srA, srB, mem2, accum, bar);

    (void)hipFuncSetAttribute((const void*)snn_persist_k,
                              hipFuncAttributeMaxDynamicSharedMemorySize, LDS_BYTES);
    snn_persist_k<<<dim3(NBLK), dim3(256), LDS_BYTES, stream>>>(
        xs, W1h, Wdh, Wrh, b1, bd, br, W2, b2,
        alpha1, ro1, alphar, ror, alpha2,
        mem1, b1s, s1, memr, bbr, srA, srB, mem2, accum, out, bar);
}
